// MyAttention_39281770889908
// MI455X (gfx1250) — matrix operand order repaired and checked by reading, not compiled, not run on hardware
//
#include <hip/hip_runtime.h>
#include <hip/hip_bf16.h>

typedef __attribute__((ext_vector_type(16))) _Float16 v16h;
typedef __attribute__((ext_vector_type(8)))  _Float16 v8h;
typedef __attribute__((ext_vector_type(8)))  float    v8f;

// Problem constants (reference: B=8, N=2048, C=512, H=W=64, sr=2 -> h=w=32)
#define BB    8
#define NN    2048
#define CC    512
#define HW32  1024      // 32*32 reduced map
#define NHEAD 8
#define HDIM  64

__device__ __forceinline__ v8f wmma_f16(v16h a, v16h b, v8f c) {
  return __builtin_amdgcn_wmma_f32_16x16x32_f16(false, a, false, b, (short)0, c,
                                                false, false);
}

// A-matrix 16x32 f16 fragment (ISA 7.12.2): lanes 0-15 row M=lane, interleaved
// K: elements {kb..kb+7, kb+16..kb+23}, kb = 0 (lanes 0-15) or 8 (lanes 16-31).
__device__ __forceinline__ v16h load_a_frag(const _Float16* rowptr, int lane) {
  const _Float16* p = rowptr + ((lane & 16) ? 8 : 0);
  v8h lo = *(const v8h*)(p);
  v8h hi = *(const v8h*)(p + 16);
  v16h r;
#pragma unroll
  for (int i = 0; i < 8; ++i) { r[i] = lo[i]; r[i + 8] = hi[i]; }
  return r;
}

// B-matrix 32x16 f16 fragment: lane n holds column n; interleaved K like A:
// {kb..kb+7, kb+16..kb+23}, kb = 0 (lanes 0-15) or 8 (lanes 16-31). colptr = &W[col*K + k0].
__device__ __forceinline__ v16h load_b_frag(const _Float16* colptr, int lane) {
  const _Float16* p = colptr + ((lane & 16) ? 8 : 0);
  v8h lo = *(const v8h*)(p);
  v8h hi = *(const v8h*)(p + 16);
  v16h r;
#pragma unroll
  for (int i = 0; i < 8; ++i) { r[i] = lo[i]; r[i + 8] = hi[i]; }
  return r;
}

// ---------------------------------------------------------------- converters
__global__ __launch_bounds__(256) void cvt_f32_to_f16_kernel(
    const float* __restrict__ in, _Float16* __restrict__ out, int n) {
  int i = blockIdx.x * 256 + threadIdx.x;
  if (i < n) out[i] = (_Float16)in[i];
}

// ------------------------------------------------------- token2map: scatter
__global__ __launch_bounds__(128) void scatter_kernel(
    const float* __restrict__ xsrc, const float* __restrict__ loc,
    const float* __restrict__ conf, float* __restrict__ featsum,
    float* __restrict__ cnt, float* __restrict__ confsum) {
  int tok = blockIdx.x;                 // 0 .. B*Nsrc-1
  int b = tok >> 11;                    // / 2048
  float lx = loc[(size_t)tok * 2 + 0];
  float ly = loc[(size_t)tok * 2 + 1];
  lx = fminf(fmaxf(lx, -1.f), 1.f);
  ly = fminf(fmaxf(ly, -1.f), 1.f);
  int x0 = (int)rintf(0.5f * (lx + 1.f) * 32.f - 0.5f);
  int y0 = (int)rintf(0.5f * (ly + 1.f) * 32.f - 0.5f);
  x0 = x0 < 0 ? 0 : (x0 > 31 ? 31 : x0);
  y0 = y0 < 0 ? 0 : (y0 > 31 ? 31 : y0);
  int cell = b * HW32 + y0 * 32 + x0;
  const float* src = xsrc + (size_t)tok * CC;
  float* dst = featsum + (size_t)cell * CC;
#pragma unroll
  for (int c = threadIdx.x; c < CC; c += 128) atomicAdd(&dst[c], src[c]);
  if (threadIdx.x == 0) {
    atomicAdd(&cnt[cell], 1.0f);
    atomicAdd(&confsum[cell], conf[tok]);
  }
}

// -------------------------------------------- token2map: normalize + fp16
__global__ __launch_bounds__(128) void finalize_kernel(
    const float* __restrict__ featsum, const float* __restrict__ cnt,
    const float* __restrict__ confsum, _Float16* __restrict__ feat16,
    float* __restrict__ confmap) {
  int cell = blockIdx.x;  // 0 .. B*HW32-1
  float c = cnt[cell];
  float inv = (c > 0.f) ? 1.f / (c + 1e-6f) : 0.f;
  const float* fs = featsum + (size_t)cell * CC;
  _Float16* fo = feat16 + (size_t)cell * CC;
  for (int i = threadIdx.x; i < CC; i += 128) fo[i] = (_Float16)(fs[i] * inv);
  if (threadIdx.x == 0) confmap[cell] = confsum[cell] * inv;
}

// ------------------- generic WMMA GEMM: D = A(MxK) * W(NxK)^T, 16x64 per wave
// Block = 4 waves covering 64 M-rows x 64 N-cols. Per K-step each wave issues
// all 10 fragment loads first, then 4 WMMAs, so the compiler can overlap
// outstanding loads with the matrix pipe (partial s_wait_loadcnt).
// mode 0: outF = acc + bias[col] (f32);  mode 1: outH = acc (f16, no bias)
__global__ __launch_bounds__(128) void gemm_wmma_kernel(
    const _Float16* __restrict__ A, const _Float16* __restrict__ W,
    const float* __restrict__ bias, float* __restrict__ outF,
    _Float16* __restrict__ outH, int M, int N, int K, int mode) {
  int wave = threadIdx.x >> 5;
  int lane = threadIdx.x & 31;
  int tm = blockIdx.y * 4 + wave;   // 16-tall M tile per wave
  int tn0 = blockIdx.x * 4;         // 4 x 16-wide N tiles per wave
  const _Float16* ap = A + (size_t)(tm * 16 + (lane & 15)) * K;
  const _Float16* wp = W + (size_t)(tn0 * 16 + (lane & 15)) * K;
  v8f acc0 = {}, acc1 = {}, acc2 = {}, acc3 = {};
#pragma unroll 2
  for (int k0 = 0; k0 < K; k0 += 32) {
    v16h af  = load_a_frag(ap + k0, lane);
    v16h bf0 = load_b_frag(wp + (size_t)0 * 16 * K + k0, lane);
    v16h bf1 = load_b_frag(wp + (size_t)1 * 16 * K + k0, lane);
    v16h bf2 = load_b_frag(wp + (size_t)2 * 16 * K + k0, lane);
    v16h bf3 = load_b_frag(wp + (size_t)3 * 16 * K + k0, lane);
    acc0 = wmma_f16(af, bf0, acc0);
    acc1 = wmma_f16(af, bf1, acc1);
    acc2 = wmma_f16(af, bf2, acc2);
    acc3 = wmma_f16(af, bf3, acc3);
  }
  v8f accs[4] = {acc0, acc1, acc2, acc3};
  int rowbase = (lane & 16) ? 8 : 0;
#pragma unroll
  for (int t = 0; t < 4; ++t) {
    int col = (tn0 + t) * 16 + (lane & 15);
    if (mode == 0) {
      float bv = bias[col];
#pragma unroll
      for (int r = 0; r < 8; ++r)
        outF[(size_t)(tm * 16 + rowbase + r) * N + col] = accs[t][r] + bv;
    } else {
#pragma unroll
      for (int r = 0; r < 8; ++r)
        outH[(size_t)(tm * 16 + rowbase + r) * N + col] = (_Float16)accs[t][r];
    }
  }
}

// ------------------------------------------------- LayerNorm + exact GELU
__global__ __launch_bounds__(256) void ln_gelu_kernel(
    const float* __restrict__ xs32, const float* __restrict__ g,
    const float* __restrict__ bparm, _Float16* __restrict__ xs16) {
  __shared__ float red[16];
  int row = blockIdx.x;
  const float* xr = xs32 + (size_t)row * CC;
  float v0 = xr[threadIdx.x];
  float v1 = xr[threadIdx.x + 256];
  float s = v0 + v1, sq = v0 * v0 + v1 * v1;
#pragma unroll
  for (int off = 1; off < 32; off <<= 1) {
    s += __shfl_xor(s, off, 32);
    sq += __shfl_xor(sq, off, 32);
  }
  if ((threadIdx.x & 31) == 0) {
    red[threadIdx.x >> 5] = s;
    red[8 + (threadIdx.x >> 5)] = sq;
  }
  __syncthreads();
  float ts = 0.f, tq = 0.f;
#pragma unroll
  for (int i = 0; i < 8; ++i) { ts += red[i]; tq += red[8 + i]; }
  float mu = ts * (1.f / CC);
  float var = tq * (1.f / CC) - mu * mu;
  float rstd = rsqrtf(var + 1e-5f);
  int c0 = threadIdx.x, c1 = threadIdx.x + 256;
  float y0 = (v0 - mu) * rstd * g[c0] + bparm[c0];
  float y1 = (v1 - mu) * rstd * g[c1] + bparm[c1];
  y0 = 0.5f * y0 * (1.f + erff(y0 * 0.70710678118654752f));
  y1 = 0.5f * y1 * (1.f + erff(y1 * 0.70710678118654752f));
  xs16[(size_t)row * CC + c0] = (_Float16)y0;
  xs16[(size_t)row * CC + c1] = (_Float16)y1;
}

// ---------------------------- V transpose: vT[(b*8+h)*64 + d][n] = v[b,n,h,d]
__global__ __launch_bounds__(256) void transpose_v_kernel(
    const _Float16* __restrict__ kv16, _Float16* __restrict__ vT16) {
  size_t i = (size_t)blockIdx.x * 256 + threadIdx.x;  // 2^22 elements
  int n = (int)(i & 1023);
  int d = (int)((i >> 10) & 63);
  int h = (int)((i >> 16) & 7);
  int b = (int)(i >> 19);
  vT16[i] = kv16[((size_t)(b * HW32 + n)) * (2 * CC) + CC + h * HDIM + d];
}

// ------------------------------------------------ flash attention (wave/tile)
// One wave: 16 q-rows x all 1024 keys for one (b, head). hd=64.
__global__ __launch_bounds__(128) void attn_kernel(
    const _Float16* __restrict__ q16, const _Float16* __restrict__ kv16,
    const _Float16* __restrict__ vT16, const float* __restrict__ confmap,
    _Float16* __restrict__ ao16) {
  __shared__ _Float16 plds[4][16 * 32];
  int wave = threadIdx.x >> 5;
  int lane = threadIdx.x & 31;
  int gw = blockIdx.x * 4 + wave;        // 0 .. 8191
  int mt = gw & 127;                     // q tile (of 128)
  int hh = (gw >> 7) & 7;                // head
  int b = gw >> 10;                      // batch
  int colL = lane & 15;
  int rowbase = (lane & 16) ? 8 : 0;
  _Float16* pl = plds[wave];

  // Q fragments (16x64 split into two 16x32 A-frags along d)
  const _Float16* qp =
      q16 + ((size_t)(b * NN + mt * 16 + colL)) * CC + hh * HDIM;
  v16h aq0 = load_a_frag(qp, lane);
  v16h aq1 = load_a_frag(qp + 32, lane);

  // V^T base for this (b, head): rows d = 0..63, each 1024 keys long
  const _Float16* vt = vT16 + ((size_t)(b * NHEAD + hh) * HDIM) * HW32;

  float mrow[8], lsum[8];
#pragma unroll
  for (int r = 0; r < 8; ++r) { mrow[r] = -3.0e38f; lsum[r] = 0.f; }
  v8f o0 = {}, o1 = {}, o2 = {}, o3 = {};

  for (int jt = 0; jt < 32; ++jt) {       // 32 keys per step
    int n0 = jt * 32;
    float conf0 = confmap[b * HW32 + n0 + colL];
    float conf1 = confmap[b * HW32 + n0 + 16 + colL];

    // S = Q * K^T  (two 16x16 tiles), contraction d=64.
    // Issue all 4 B-fragment loads first, then the WMMAs (overlap).
    const _Float16* kp0 =
        kv16 + ((size_t)(b * HW32 + n0 + colL)) * (2 * CC) + hh * HDIM;
    const _Float16* kp1 = kp0 + (size_t)16 * (2 * CC);
    v16h bk0 = load_b_frag(kp0, lane);
    v16h bk1 = load_b_frag(kp0 + 32, lane);
    v16h bk2 = load_b_frag(kp1, lane);
    v16h bk3 = load_b_frag(kp1 + 32, lane);
    v8f s0 = {}, s1 = {};
    s0 = wmma_f16(aq0, bk0, s0);
    s0 = wmma_f16(aq1, bk1, s0);
    s1 = wmma_f16(aq0, bk2, s1);
    s1 = wmma_f16(aq1, bk3, s1);

    // online softmax over the 32 new columns
    float p0[8], p1[8], alpha[8];
#pragma unroll
    for (int r = 0; r < 8; ++r) {
      float v0 = s0[r] * 0.125f + conf0;
      float v1 = s1[r] * 0.125f + conf1;
      float t = fmaxf(v0, v1);
#pragma unroll
      for (int off = 1; off < 16; off <<= 1)
        t = fmaxf(t, __shfl_xor(t, off, 16));
      float mn = fmaxf(mrow[r], t);
      float al = __expf(mrow[r] - mn);
      mrow[r] = mn;
      alpha[r] = al;
      float e0 = __expf(v0 - mn);
      float e1 = __expf(v1 - mn);
      float ls = e0 + e1;
#pragma unroll
      for (int off = 1; off < 16; off <<= 1) ls += __shfl_xor(ls, off, 16);
      lsum[r] = lsum[r] * al + ls;
      p0[r] = e0;
      p1[r] = e1;
    }

    // Prefetch V^T fragments for this key block (contiguous over keys)
    v16h bv0 = load_b_frag(vt + (size_t)(0 * 16 + colL) * HW32 + n0, lane);
    v16h bv1 = load_b_frag(vt + (size_t)(1 * 16 + colL) * HW32 + n0, lane);
    v16h bv2 = load_b_frag(vt + (size_t)(2 * 16 + colL) * HW32 + n0, lane);
    v16h bv3 = load_b_frag(vt + (size_t)(3 * 16 + colL) * HW32 + n0, lane);

    // Rescale accumulators by alpha
#pragma unroll
    for (int r = 0; r < 8; ++r) {
      o0[r] *= alpha[r];
      o1[r] *= alpha[r];
      o2[r] *= alpha[r];
      o3[r] *= alpha[r];
    }

    // stage P (16x32) in LDS; D-layout store is already row-major
#pragma unroll
    for (int r = 0; r < 8; ++r) {
      pl[(rowbase + r) * 32 + colL] = (_Float16)p0[r];
      pl[(rowbase + r) * 32 + 16 + colL] = (_Float16)p1[r];
    }
    asm volatile("s_wait_dscnt 0" ::: "memory");

    // P as A-fragment (16x32)
    v16h apf = load_a_frag(pl + colL * 32, lane);

    // O += P * V
    o0 = wmma_f16(apf, bv0, o0);
    o1 = wmma_f16(apf, bv1, o1);
    o2 = wmma_f16(apf, bv2, o2);
    o3 = wmma_f16(apf, bv3, o3);
    asm volatile("s_wait_dscnt 0" ::: "memory");
  }

  // normalize and store (f16 for proj GEMM)
  v8f os[4] = {o0, o1, o2, o3};
#pragma unroll
  for (int t = 0; t < 4; ++t) {
#pragma unroll
    for (int r = 0; r < 8; ++r) {
      int m = mt * 16 + rowbase + r;
      ao16[((size_t)(b * NN + m)) * CC + hh * HDIM + t * 16 + colL] =
          (_Float16)(os[t][r] / lsum[r]);
    }
  }
}

// ======================================================================
extern "C" void kernel_launch(void* const* d_in, const int* in_sizes, int n_in,
                              void* d_out, int out_size, void* d_ws,
                              size_t ws_size, hipStream_t stream) {
  const float* x      = (const float*)d_in[0];
  const float* xsrc   = (const float*)d_in[1];
  const float* loc    = (const float*)d_in[2];
  const float* conf   = (const float*)d_in[3];
  const float* q_w    = (const float*)d_in[4];
  const float* kv_w   = (const float*)d_in[5];
  const float* sr_w   = (const float*)d_in[6];
  const float* sr_b   = (const float*)d_in[7];
  const float* ln_g   = (const float*)d_in[8];
  const float* ln_b   = (const float*)d_in[9];
  const float* proj_w = (const float*)d_in[10];
  const float* proj_b = (const float*)d_in[11];
  float* out = (float*)d_out;

  const size_t MB = 1ull << 20;
  char* ws = (char*)d_ws;
  float*    featsum = (float*)(ws);                      // 16 MB, reused as xs32
  float*    cnt     = (float*)(ws + 16 * MB);            // 32 KB
  float*    confsum = (float*)(ws + 16 * MB + 32 * 1024);
  float*    confmap = (float*)(ws + 16 * MB + 64 * 1024);
  _Float16* feat16  = (_Float16*)(ws + 17 * MB);         // 8 MB
  _Float16* x16     = (_Float16*)(ws + 25 * MB);         // 16 MB (dead after q GEMM)
  _Float16* vT16    = (_Float16*)(ws + 25 * MB);         // 8 MB, reuses x16 slot
  _Float16* q16     = (_Float16*)(ws + 41 * MB);         // 16 MB
  _Float16* xs16    = (_Float16*)(ws + 57 * MB);         // 8 MB
  _Float16* kv16    = (_Float16*)(ws + 65 * MB);         // 16 MB
  _Float16* ao16    = (_Float16*)(ws + 81 * MB);         // 16 MB
  _Float16* qw16    = (_Float16*)(ws + 97 * MB);         // 512 KB
  _Float16* srw16   = (_Float16*)(ws + 97 * MB + 512 * 1024);
  _Float16* kvw16   = (_Float16*)(ws + 98 * MB);         // 1 MB
  _Float16* projw16 = (_Float16*)(ws + 99 * MB);         // 512 KB

  // zero scatter accumulators (featsum + cnt + confsum)
  hipMemsetAsync(ws, 0, 16 * MB + 64 * 1024, stream);

  // fp16 conversions
  cvt_f32_to_f16_kernel<<<(BB * NN * CC + 255) / 256, 256, 0, stream>>>(
      x, x16, BB * NN * CC);
  cvt_f32_to_f16_kernel<<<(CC * CC + 255) / 256, 256, 0, stream>>>(q_w, qw16,
                                                                   CC * CC);
  cvt_f32_to_f16_kernel<<<(CC * CC + 255) / 256, 256, 0, stream>>>(sr_w, srw16,
                                                                   CC * CC);
  cvt_f32_to_f16_kernel<<<(2 * CC * CC + 255) / 256, 256, 0, stream>>>(
      kv_w, kvw16, 2 * CC * CC);
  cvt_f32_to_f16_kernel<<<(CC * CC + 255) / 256, 256, 0, stream>>>(
      proj_w, projw16, CC * CC);

  // token2map scatter + normalize
  scatter_kernel<<<BB * NN, 128, 0, stream>>>(xsrc, loc, conf, featsum, cnt,
                                              confsum);
  finalize_kernel<<<BB * HW32, 128, 0, stream>>>(featsum, cnt, confsum, feat16,
                                                 confmap);

  // xs = feat @ sr_w^T + sr_b   (f32 out into featsum region)
  {
    dim3 grid(CC / 64, (BB * HW32) / 64);
    gemm_wmma_kernel<<<grid, 128, 0, stream>>>(feat16, srw16, sr_b, featsum,
                                               nullptr, BB * HW32, CC, CC, 0);
  }
  // LayerNorm + GELU -> xs16
  ln_gelu_kernel<<<BB * HW32, 256, 0, stream>>>(featsum, ln_g, ln_b, xs16);

  // kv = xs @ kv_w^T  (f16 out)
  {
    dim3 grid((2 * CC) / 64, (BB * HW32) / 64);
    gemm_wmma_kernel<<<grid, 128, 0, stream>>>(xs16, kvw16, nullptr, nullptr,
                                               kv16, BB * HW32, 2 * CC, CC, 1);
  }
  // q = x @ q_w^T  (f16 out) — last consumer of x16
  {
    dim3 grid(CC / 64, (BB * NN) / 64);
    gemm_wmma_kernel<<<grid, 128, 0, stream>>>(x16, qw16, nullptr, nullptr, q16,
                                               BB * NN, CC, CC, 1);
  }
  // V^T for coalesced P*V fragments (into the now-dead x16 slot)
  transpose_v_kernel<<<(BB * NHEAD * HDIM * HW32) / 256, 256, 0, stream>>>(
      kv16, vT16);
  // attention
  attn_kernel<<<(BB * NHEAD * (NN / 16)) / 4, 128, 0, stream>>>(
      q16, kv16, vT16, confmap, ao16);
  // out = ao @ proj_w^T + proj_b (f32)
  {
    dim3 grid(CC / 64, (BB * NN) / 64);
    gemm_wmma_kernel<<<grid, 128, 0, stream>>>(ao16, projw16, proj_b, out,
                                               nullptr, BB * NN, CC, CC, 0);
  }
}
